// TextEncoder_12171937317582
// MI455X (gfx1250) — compile-verified, matrix operand order checked by reading, not run on hardware
//
#include <hip/hip_runtime.h>
#include <hip/hip_bf16.h>
#include <math.h>

// ---------------- model constants ----------------
#define BB   4
#define TT   768
#define VV   512
#define DD   512
#define FF   2048
#define HH   8
#define LLAYERS 6
#define KK   3
#define WSZ  4
#define OUTC 192
#define KCH  64          // D / H

typedef __bf16 bf16_t;
typedef __attribute__((ext_vector_type(16))) __bf16 v16bf;
typedef __attribute__((ext_vector_type(8)))  __bf16 v8bf;
typedef __attribute__((ext_vector_type(8)))  float  v8f;

// ---------------------------------------------------------------------------
// WMMA fragment loaders (CDNA5 16-bit 16x32 A/B layout, wave32):
//   lanes 0-15  : row M = lane,      v0..3 = K 0..7,  v4..7 = K 16..23
//   lanes 16-31 : row M = lane-16,   v0..3 = K 8..15, v4..7 = K 24..31
// Each lane does two contiguous 16-byte loads.
// ---------------------------------------------------------------------------
__device__ __forceinline__ v16bf frag_ld(const bf16_t* __restrict__ base,
                                         int ld, int row0, int k0, int lane) {
  int r    = row0 + (lane & 15);
  int half = (lane >> 4) << 3;         // 0 or 8
  const bf16_t* p = base + (long)r * ld + k0 + half;
  v16bf f;
  ((v8bf*)&f)[0] = *(const v8bf*)(p);
  ((v8bf*)&f)[1] = *(const v8bf*)(p + 16);
  return f;
}

__device__ __forceinline__ v16bf frag_ld_guard(const bf16_t* __restrict__ base,
                                               int ld, int row0, int k0,
                                               int lane, int nrows) {
  int r    = row0 + (lane & 15);
  int half = (lane >> 4) << 3;
  v16bf f;
  if ((unsigned)r < (unsigned)nrows) {
    const bf16_t* p = base + (long)r * ld + k0 + half;
    ((v8bf*)&f)[0] = *(const v8bf*)(p);
    ((v8bf*)&f)[1] = *(const v8bf*)(p + 16);
  } else {
    for (int i = 0; i < 16; ++i) f[i] = (__bf16)0.0f;
  }
  return f;
}

#define WMMA_BF16(A, Bf, C) \
  __builtin_amdgcn_wmma_f32_16x16x32_bf16(false, (A), false, (Bf), (short)0, (C), false, false)

// ---------------------------------------------------------------------------
// Epilogue: write one 16x16 f32 C tile with fused bias / scale / ReLU / mask.
// MODE 0: f32 token-major out[n*ldo+m]; MODE 1: bf16 token-major (single 16-B
// store per lane); MODE 2: bf16 channel-major out[m*ldo+n] (scatter).
// ---------------------------------------------------------------------------
template <int MODE, bool RELU>
__device__ __forceinline__ void epilogue_tile(
    v8f c, int mBase, int nBase, int lane,
    const float* __restrict__ bias, const float* __restrict__ maskRow,
    float scaleOut, float* __restrict__ outF, bf16_t* __restrict__ outB,
    long ob, int ldo) {
  int n  = lane & 15;
  int mo = (lane >> 4) << 3;
  int ng = nBase + n;
  float mk = maskRow ? maskRow[ng] : 1.0f;
  float vals[8];
  for (int i = 0; i < 8; ++i) {
    float v = c[i];
    if (bias) v += bias[mBase + mo + i];
    v *= scaleOut;
    if (RELU) v = v > 0.0f ? v : 0.0f;
    vals[i] = v * mk;
  }
  if (MODE == 0) {
    float* p = outF + ob + (long)ng * ldo + mBase + mo;
    for (int i = 0; i < 8; ++i) p[i] = vals[i];
  } else if (MODE == 1) {
    v8bf o;
    for (int i = 0; i < 8; ++i) o[i] = (bf16_t)vals[i];
    *(v8bf*)(outB + ob + (long)ng * ldo + mBase + mo) = o;
  } else {
    for (int i = 0; i < 8; ++i)
      outB[ob + (long)(mBase + mo + i) * ldo + ng] = (bf16_t)vals[i];
  }
}

// ---------------------------------------------------------------------------
// pack fp32 weights -> bf16, arbitrary (strideM, strideC, offset) gather.
// ---------------------------------------------------------------------------
__global__ void pack_w(const float* __restrict__ src, bf16_t* __restrict__ dst,
                       int M, int Kd, int sm, int sc, int off) {
  long i = (long)blockIdx.x * 256 + threadIdx.x;
  long tot = (long)M * Kd;
  if (i >= tot) return;
  int m = (int)(i / Kd), c = (int)(i % Kd);
  dst[i] = (bf16_t)src[(long)m * sm + (long)c * sc + off];
}

// ---------------------------------------------------------------------------
// Embedding: h[b,t,d] = (emb[x] + (t_id!=0)*emb_t[t_id]) * sqrt(D) * mask
// ---------------------------------------------------------------------------
__global__ void embed_kernel(const int* __restrict__ x, const int* __restrict__ tids,
                             const int* __restrict__ lens,
                             const float* __restrict__ emb, const float* __restrict__ embt,
                             float* __restrict__ hF, bf16_t* __restrict__ hB,
                             float* __restrict__ maskF, float* __restrict__ outMask) {
  int t = blockIdx.x, b = blockIdx.y;
  int len = lens[b];
  float mk = (t < len) ? 1.0f : 0.0f;
  if (threadIdx.x == 0) { maskF[b * TT + t] = mk; outMask[b * TT + t] = mk; }
  int tok = x[b * TT + t];
  int ty  = tids[b * TT + t];
  const float sq = 22.62741699796952f;   // sqrt(512)
  long o = ((long)b * TT + t) * DD;
  for (int d = threadIdx.x; d < DD; d += blockDim.x) {
    float v = emb[(long)tok * DD + d];
    if (ty != 0) v += embt[(long)ty * DD + d];
    v *= sq * mk;
    hF[o + d] = v;
    hB[o + d] = (bf16_t)v;
  }
}

// ---------------------------------------------------------------------------
// Generic WMMA GEMM, register-blocked 32x32 per wave (2 M-tiles x 2 N-tiles):
//   out[m,n] = sum_taps sum_c W_tap[m,c] * X[n + tap - taps/2, c]
// 4 WMMAs per K-step on 4 fragment loads -> 2x arithmetic intensity.
// ---------------------------------------------------------------------------
template <int MODE, bool RELU>
__global__ __launch_bounds__(128) void gemm_tok(
    const bf16_t* __restrict__ W, const bf16_t* __restrict__ X,
    const float* __restrict__ bias, const float* __restrict__ maskBT,
    float* __restrict__ outF, bf16_t* __restrict__ outB,
    int M, int N, int Kd, int ldx, int taps,
    long xBS, long oBS, int ldo, float scaleOut) {
  int lane  = threadIdx.x & 31;
  int wave  = threadIdx.x >> 5;
  int nP = blockIdx.y * 4 + wave;          // 32 tokens per wave
  if (nP * 32 >= N) return;
  int mP = blockIdx.x;                     // 32 output channels per wave
  int b  = blockIdx.z;
  const bf16_t* Xb = X + (long)b * xBS;

  v8f c00, c01, c10, c11;
  for (int i = 0; i < 8; ++i) { c00[i] = 0.f; c01[i] = 0.f; c10[i] = 0.f; c11[i] = 0.f; }

  for (int tap = 0; tap < taps; ++tap) {
    int sh = tap - (taps >> 1);            // conv shift: -1,0,1 (taps=3)
    const bf16_t* Wt = W + (long)tap * M * Kd;
    for (int k0 = 0; k0 < Kd; k0 += 32) {
      if (k0 + 64 < Kd) {                  // stream-ahead hint into WGP$/L2
        __builtin_prefetch(Wt + (long)(mP * 32 + (lane & 15)) * Kd + k0 + 64, 0, 1);
        __builtin_prefetch(Wt + (long)(mP * 32 + 16 + (lane & 15)) * Kd + k0 + 64, 0, 1);
      }
      v16bf a0 = frag_ld(Wt, Kd, mP * 32,      k0, lane);
      v16bf a1 = frag_ld(Wt, Kd, mP * 32 + 16, k0, lane);
      v16bf b0 = frag_ld_guard(Xb, ldx, nP * 32 + sh,      k0, lane, N);
      v16bf b1 = frag_ld_guard(Xb, ldx, nP * 32 + 16 + sh, k0, lane, N);
      c00 = WMMA_BF16(a0, b0, c00);
      c01 = WMMA_BF16(a0, b1, c01);
      c10 = WMMA_BF16(a1, b0, c10);
      c11 = WMMA_BF16(a1, b1, c11);
    }
  }

  const float* maskRow = maskBT ? maskBT + b * N : nullptr;
  long ob = (long)b * oBS;
  epilogue_tile<MODE, RELU>(c00, mP * 32,      nP * 32,      lane, bias, maskRow, scaleOut, outF, outB, ob, ldo);
  epilogue_tile<MODE, RELU>(c01, mP * 32,      nP * 32 + 16, lane, bias, maskRow, scaleOut, outF, outB, ob, ldo);
  epilogue_tile<MODE, RELU>(c10, mP * 32 + 16, nP * 32,      lane, bias, maskRow, scaleOut, outF, outB, ob, ldo);
  epilogue_tile<MODE, RELU>(c11, mP * 32 + 16, nP * 32 + 16, lane, bias, maskRow, scaleOut, outF, outB, ob, ldo);
}

// ---------------------------------------------------------------------------
// scores[b,h,t,s] = q[t,:].k[s,:] (q pre-scaled).  32x32 per wave.
// A = k rows (M = s), B = q rows (N = t) -> s-contiguous f32 rows out.
// ---------------------------------------------------------------------------
__global__ __launch_bounds__(128) void gemm_scores(const bf16_t* __restrict__ qB,
                                                   const bf16_t* __restrict__ kB,
                                                   float* __restrict__ scores) {
  int lane = threadIdx.x & 31;
  int wave = threadIdx.x >> 5;
  int tP = blockIdx.y * 4 + wave;      // t pair
  int sP = blockIdx.x;                 // s pair
  int z = blockIdx.z; int b = z >> 3, h = z & 7;
  const bf16_t* qb = qB + (long)b * TT * DD + h * KCH;
  const bf16_t* kb = kB + (long)b * TT * DD + h * KCH;

  v8f c00, c01, c10, c11;
  for (int i = 0; i < 8; ++i) { c00[i] = 0.f; c01[i] = 0.f; c10[i] = 0.f; c11[i] = 0.f; }
  for (int k0 = 0; k0 < KCH; k0 += 32) {
    v16bf a0 = frag_ld(kb, DD, sP * 32,      k0, lane);
    v16bf a1 = frag_ld(kb, DD, sP * 32 + 16, k0, lane);
    v16bf b0 = frag_ld(qb, DD, tP * 32,      k0, lane);
    v16bf b1 = frag_ld(qb, DD, tP * 32 + 16, k0, lane);
    c00 = WMMA_BF16(a0, b0, c00);
    c01 = WMMA_BF16(a0, b1, c01);
    c10 = WMMA_BF16(a1, b0, c10);
    c11 = WMMA_BF16(a1, b1, c11);
  }
  float* sc = scores + (long)z * TT * TT;
  int n = lane & 15, mo = (lane >> 4) << 3;
  for (int i = 0; i < 8; ++i) {
    sc[(long)(tP * 32 + n) * TT      + sP * 32      + mo + i] = c00[i];
    sc[(long)(tP * 32 + 16 + n) * TT + sP * 32      + mo + i] = c01[i];
    sc[(long)(tP * 32 + n) * TT      + sP * 32 + 16 + mo + i] = c10[i];
    sc[(long)(tP * 32 + 16 + n) * TT + sP * 32 + 16 + mo + i] = c11[i];
  }
}

// ---------------------------------------------------------------------------
// attn[b,t,h*64+d] = sum_s p[t,s] * v[s,d].  32x32 per wave, K = 768.
// A = p rows (M = t), B = v channel-major rows (N = d).
// ---------------------------------------------------------------------------
__global__ __launch_bounds__(64) void gemm_pv(const bf16_t* __restrict__ pB,
                                              const bf16_t* __restrict__ vch,
                                              float* __restrict__ attnF) {
  int lane = threadIdx.x & 31;
  int dP   = threadIdx.x >> 5;         // d pair: 0..1 (64 channels)
  int tP   = blockIdx.x;               // t pair
  int z = blockIdx.z; int b = z >> 3, h = z & 7;
  const bf16_t* pb = pB  + (long)z * TT * TT;
  const bf16_t* vb = vch + (long)b * DD * TT + (long)h * KCH * TT;

  v8f c00, c01, c10, c11;
  for (int i = 0; i < 8; ++i) { c00[i] = 0.f; c01[i] = 0.f; c10[i] = 0.f; c11[i] = 0.f; }
  for (int k0 = 0; k0 < TT; k0 += 32) {
    if (k0 + 64 < TT)
      __builtin_prefetch(pb + (long)(tP * 32 + (lane & 15)) * TT + k0 + 64, 0, 1);
    v16bf a0 = frag_ld(pb, TT, tP * 32,      k0, lane);
    v16bf a1 = frag_ld(pb, TT, tP * 32 + 16, k0, lane);
    v16bf b0 = frag_ld(vb, TT, dP * 32,      k0, lane);
    v16bf b1 = frag_ld(vb, TT, dP * 32 + 16, k0, lane);
    c00 = WMMA_BF16(a0, b0, c00);
    c01 = WMMA_BF16(a0, b1, c01);
    c10 = WMMA_BF16(a1, b0, c10);
    c11 = WMMA_BF16(a1, b1, c11);
  }
  float* ob = attnF + (long)b * TT * DD + h * KCH;
  int n = lane & 15, mo = (lane >> 4) << 3;
  for (int i = 0; i < 8; ++i) {
    ob[(long)(tP * 32 + mo + i) * DD      + dP * 32      + n] = c00[i];
    ob[(long)(tP * 32 + mo + i) * DD      + dP * 32 + 16 + n] = c01[i];
    ob[(long)(tP * 32 + 16 + mo + i) * DD + dP * 32      + n] = c10[i];
    ob[(long)(tP * 32 + 16 + mo + i) * DD + dP * 32 + 16 + n] = c11[i];
  }
}

// ---------------------------------------------------------------------------
// Relative-position score band: scores[t, t+j-4] += q[t] . rel_k[j], j=0..8
// ---------------------------------------------------------------------------
__global__ void rel_score_kernel(const bf16_t* __restrict__ qB,
                                 const float* __restrict__ relk,
                                 float* __restrict__ scores) {
  int t = blockIdx.x, h = blockIdx.y, b = blockIdx.z;
  int lane = threadIdx.x;              // 32 lanes, 2 channels each
  const bf16_t* q = qB + ((long)b * TT + t) * DD + h * KCH;
  float q0 = (float)q[lane], q1 = (float)q[lane + 32];
  float* sc = scores + ((long)(b * HH + h) * TT + t) * TT;
  for (int j = 0; j < 2 * WSZ + 1; ++j) {
    int s = t + j - WSZ;
    float d = q0 * relk[j * KCH + lane] + q1 * relk[j * KCH + lane + 32];
    for (int m = 16; m; m >>= 1) d += __shfl_xor(d, m, 32);
    if (lane == 0 && s >= 0 && s < TT) sc[s] += d;
  }
}

// ---------------------------------------------------------------------------
// Masked softmax over s; bf16 probabilities out.
// ---------------------------------------------------------------------------
__global__ void softmax_kernel(const float* __restrict__ scores,
                               bf16_t* __restrict__ pB,
                               const int* __restrict__ lens) {
  int t = blockIdx.x, h = blockIdx.y, b = blockIdx.z;
  int len = lens[b];
  const float* row = scores + ((long)(b * HH + h) * TT + t) * TT;
  bf16_t* prow = pB + ((long)(b * HH + h) * TT + t) * TT;
  __shared__ float red[256];
  int tid = threadIdx.x;
  bool tvalid = t < len;
  float vals[3];
  float mx = -3.4e38f;
  for (int i = 0; i < 3; ++i) {
    int s = tid + i * 256;
    float v = (tvalid && s < len) ? row[s] : -10000.0f;
    vals[i] = v;
    mx = fmaxf(mx, v);
  }
  red[tid] = mx; __syncthreads();
  for (int st = 128; st; st >>= 1) {
    if (tid < st) red[tid] = fmaxf(red[tid], red[tid + st]);
    __syncthreads();
  }
  mx = red[0]; __syncthreads();
  float sum = 0.0f;
  for (int i = 0; i < 3; ++i) { vals[i] = __expf(vals[i] - mx); sum += vals[i]; }
  red[tid] = sum; __syncthreads();
  for (int st = 128; st; st >>= 1) {
    if (tid < st) red[tid] += red[tid + st];
    __syncthreads();
  }
  float inv = 1.0f / red[0];
  for (int i = 0; i < 3; ++i)
    prow[tid + i * 256] = (bf16_t)(vals[i] * inv);
}

// ---------------------------------------------------------------------------
// Relative-position value band: attn[t,d] += sum_j p[t, t+j-4] * rel_v[j,d]
// ---------------------------------------------------------------------------
__global__ void rel_v_kernel(const bf16_t* __restrict__ pB,
                             const float* __restrict__ relv,
                             float* __restrict__ attnF) {
  int t = blockIdx.x, h = blockIdx.y, b = blockIdx.z;
  int d = threadIdx.x;                 // 64
  const bf16_t* prow = pB + ((long)(b * HH + h) * TT + t) * TT;
  float acc = 0.0f;
  for (int j = 0; j < 2 * WSZ + 1; ++j) {
    int s = t + j - WSZ;
    if (s >= 0 && s < TT) acc += (float)prow[s] * relv[j * KCH + d];
  }
  attnF[((long)b * TT + t) * DD + h * KCH + d] += acc;
}

// f32 -> bf16 copy (attn output becomes GEMM input)
__global__ void f2b_kernel(const float* __restrict__ src,
                           bf16_t* __restrict__ dst, long n) {
  long i = (long)blockIdx.x * 256 + threadIdx.x;
  if (i < n) dst[i] = (bf16_t)src[i];
}

// ---------------------------------------------------------------------------
// LayerNorm over channels: h = g*(x-mu)*rsqrt(var+1e-5)+b, x = h_prev + y
// ---------------------------------------------------------------------------
__global__ void ln_kernel(float* __restrict__ hF, const float* __restrict__ yF,
                          const float* __restrict__ g, const float* __restrict__ bta,
                          const float* __restrict__ maskF, bf16_t* __restrict__ hB) {
  int t = blockIdx.x, b = blockIdx.y, tid = threadIdx.x;   // 256 threads
  long o = ((long)b * TT + t) * DD;
  float x0 = hF[o + tid]       + yF[o + tid];
  float x1 = hF[o + tid + 256] + yF[o + tid + 256];
  __shared__ float s1[256], s2[256];
  s1[tid] = x0 + x1;
  s2[tid] = x0 * x0 + x1 * x1;
  __syncthreads();
  for (int st = 128; st; st >>= 1) {
    if (tid < st) { s1[tid] += s1[tid + st]; s2[tid] += s2[tid + st]; }
    __syncthreads();
  }
  float mu  = s1[0] * (1.0f / DD);
  float var = s2[0] * (1.0f / DD) - mu * mu;
  float rs  = rsqrtf(var + 1e-5f);
  float mk  = maskF[b * TT + t];
  float o0 = g[tid]       * (x0 - mu) * rs + bta[tid];
  float o1 = g[tid + 256] * (x1 - mu) * rs + bta[tid + 256];
  hF[o + tid]       = o0;
  hF[o + tid + 256] = o1;
  hB[o + tid]       = (bf16_t)(o0 * mk);
  hB[o + tid + 256] = (bf16_t)(o1 * mk);
}

// final h output: (B,T,D) f32 -> (B,D,T) masked
__global__ void final_h_kernel(const float* __restrict__ hF,
                               const float* __restrict__ maskF,
                               float* __restrict__ out) {
  int t = blockIdx.x, b = blockIdx.y;
  float mk = maskF[b * TT + t];
  for (int d = threadIdx.x; d < DD; d += blockDim.x)
    out[((long)(b * DD + d)) * TT + t] = hF[((long)b * TT + t) * DD + d] * mk;
}

// split stats (B,T,384) token-major (already masked) -> m / logs (B,192,T)
__global__ void write_stats_kernel(const float* __restrict__ statsF,
                                   float* __restrict__ outM,
                                   float* __restrict__ outL) {
  int t = blockIdx.x, b = blockIdx.y, o = threadIdx.x;     // 192 threads
  const float* s = statsF + ((long)b * TT + t) * (2 * OUTC);
  outM[((long)(b * OUTC + o)) * TT + t] = s[o];
  outL[((long)(b * OUTC + o)) * TT + t] = s[OUTC + o];
}

// ---------------------------------------------------------------------------
// host driver
// ---------------------------------------------------------------------------
extern "C" void kernel_launch(void* const* d_in, const int* in_sizes, int n_in,
                              void* d_out, int out_size, void* d_ws, size_t ws_size,
                              hipStream_t stream) {
  (void)in_sizes; (void)n_in; (void)out_size; (void)ws_size;

  const int*   x     = (const int*)d_in[0];
  const int*   tids  = (const int*)d_in[1];
  const int*   lens  = (const int*)d_in[2];
  const float* emb   = (const float*)d_in[3];
  const float* embt  = (const float*)d_in[4];
  const int LBASE = 5;
  const float* projw = (const float*)d_in[LBASE + LLAYERS * 18];
  const float* projb = (const float*)d_in[LBASE + LLAYERS * 18 + 1];

  // bump allocator over workspace (256-B aligned)
  char* ws = (char*)d_ws;
  size_t off = 0;
  auto alloc = [&](size_t bytes) -> void* {
    off = (off + 255) & ~(size_t)255;
    void* p = ws + off;
    off += bytes;
    return p;
  };

  const long BTD = (long)BB * TT * DD;
  bf16_t* wp0  = (bf16_t*)alloc((size_t)DD * DD * 2);                 // wq / proj
  bf16_t* wp1  = (bf16_t*)alloc((size_t)DD * DD * 2);                 // wk
  bf16_t* wp2  = (bf16_t*)alloc((size_t)DD * DD * 2);                 // wv
  bf16_t* wp3  = (bf16_t*)alloc((size_t)DD * DD * 2);                 // wo
  bf16_t* wpF1 = (bf16_t*)alloc((size_t)KK * FF * DD * 2);            // w1 taps
  bf16_t* wpF2 = (bf16_t*)alloc((size_t)KK * DD * FF * 2);            // w2 taps
  float*  hF    = (float*) alloc(BTD * 4);
  bf16_t* hB    = (bf16_t*)alloc(BTD * 2);
  bf16_t* qB    = (bf16_t*)alloc(BTD * 2);
  bf16_t* kB    = (bf16_t*)alloc(BTD * 2);
  bf16_t* vch   = (bf16_t*)alloc(BTD * 2);
  float*  attnF = (float*) alloc(BTD * 4);
  bf16_t* attnB = (bf16_t*)alloc(BTD * 2);
  bf16_t* y1B   = (bf16_t*)alloc((size_t)BB * TT * FF * 2);
  float*  yF    = (float*) alloc(BTD * 4);                            // also statsF
  float*  scores= (float*) alloc((size_t)BB * HH * TT * TT * 4);
  bf16_t* pB    = (bf16_t*)alloc((size_t)BB * HH * TT * TT * 2);
  float*  maskF = (float*) alloc((size_t)BB * TT * 4);

  float* out     = (float*)d_out;
  float* outH    = out;                                   // B*D*T
  float* outM    = outH + (long)BB * DD * TT;             // B*192*T
  float* outL    = outM + (long)BB * OUTC * TT;           // B*192*T
  float* outMask = outL + (long)BB * OUTC * TT;           // B*T

  dim3 blk128(128), blk256(256);
  dim3 gTok(TT, BB);
  auto packGrid = [](long n) { return dim3((unsigned)((n + 255) / 256)); };
  const int NPB = TT / 32 / 4;   // 6 n-pair blocks (4 waves * 32 tokens each)

  // ---- embedding + mask ----
  embed_kernel<<<gTok, blk256, 0, stream>>>(x, tids, lens, emb, embt,
                                            hF, hB, maskF, outMask);

  for (int l = 0; l < LLAYERS; ++l) {
    const float* wq = (const float*)d_in[LBASE + l * 18 + 0];
    const float* bq = (const float*)d_in[LBASE + l * 18 + 1];
    const float* wk = (const float*)d_in[LBASE + l * 18 + 2];
    const float* bk = (const float*)d_in[LBASE + l * 18 + 3];
    const float* wv = (const float*)d_in[LBASE + l * 18 + 4];
    const float* bv = (const float*)d_in[LBASE + l * 18 + 5];
    const float* wo = (const float*)d_in[LBASE + l * 18 + 6];
    const float* bo = (const float*)d_in[LBASE + l * 18 + 7];
    const float* relk = (const float*)d_in[LBASE + l * 18 + 8];
    const float* relv = (const float*)d_in[LBASE + l * 18 + 9];
    const float* g1 = (const float*)d_in[LBASE + l * 18 + 10];
    const float* b1 = (const float*)d_in[LBASE + l * 18 + 11];
    const float* w1 = (const float*)d_in[LBASE + l * 18 + 12];
    const float* bf1= (const float*)d_in[LBASE + l * 18 + 13];
    const float* w2 = (const float*)d_in[LBASE + l * 18 + 14];
    const float* bf2= (const float*)d_in[LBASE + l * 18 + 15];
    const float* g2 = (const float*)d_in[LBASE + l * 18 + 16];
    const float* b2 = (const float*)d_in[LBASE + l * 18 + 17];

    // ---- pack weights to bf16 (L2-resident panels) ----
    pack_w<<<packGrid((long)DD * DD), blk256, 0, stream>>>(wq, wp0, DD, DD, DD, 1, 0);
    pack_w<<<packGrid((long)DD * DD), blk256, 0, stream>>>(wk, wp1, DD, DD, DD, 1, 0);
    pack_w<<<packGrid((long)DD * DD), blk256, 0, stream>>>(wv, wp2, DD, DD, DD, 1, 0);
    pack_w<<<packGrid((long)DD * DD), blk256, 0, stream>>>(wo, wp3, DD, DD, DD, 1, 0);
    for (int k = 0; k < KK; ++k) {
      pack_w<<<packGrid((long)FF * DD), blk256, 0, stream>>>(
          w1, wpF1 + (long)k * FF * DD, FF, DD, DD * KK, KK, k);
      pack_w<<<packGrid((long)DD * FF), blk256, 0, stream>>>(
          w2, wpF2 + (long)k * DD * FF, DD, FF, FF * KK, KK, k);
    }

    // ---- Q, K (token-major bf16), V (channel-major bf16) ----
    dim3 gProj(DD / 32, NPB, BB);
    gemm_tok<1, false><<<gProj, blk128, 0, stream>>>(
        wp0, hB, bq, nullptr, nullptr, qB, DD, TT, DD, DD, 1,
        (long)TT * DD, (long)TT * DD, DD, 0.125f);           // q / sqrt(64)
    gemm_tok<1, false><<<gProj, blk128, 0, stream>>>(
        wp1, hB, bk, nullptr, nullptr, kB, DD, TT, DD, DD, 1,
        (long)TT * DD, (long)TT * DD, DD, 1.0f);
    gemm_tok<2, false><<<gProj, blk128, 0, stream>>>(
        wp2, hB, bv, nullptr, nullptr, vch, DD, TT, DD, DD, 1,
        (long)TT * DD, (long)DD * TT, TT, 1.0f);

    // ---- attention scores + relative band + softmax ----
    gemm_scores<<<dim3(TT / 32, NPB, BB * HH), blk128, 0, stream>>>(qB, kB, scores);
    rel_score_kernel<<<dim3(TT, HH, BB), dim3(32), 0, stream>>>(qB, relk, scores);
    softmax_kernel<<<dim3(TT, HH, BB), blk256, 0, stream>>>(scores, pB, lens);

    // ---- p @ v + relative-value band ----
    gemm_pv<<<dim3(TT / 32, 1, BB * HH), dim3(64), 0, stream>>>(pB, vch, attnF);
    rel_v_kernel<<<dim3(TT, HH, BB), dim3(KCH), 0, stream>>>(pB, relv, attnF);
    f2b_kernel<<<packGrid(BTD), blk256, 0, stream>>>(attnF, attnB, BTD);

    // ---- output projection + residual LN1 ----
    gemm_tok<0, false><<<gProj, blk128, 0, stream>>>(
        wp3, attnB, bo, nullptr, yF, nullptr, DD, TT, DD, DD, 1,
        (long)TT * DD, (long)TT * DD, DD, 1.0f);
    ln_kernel<<<gTok, blk256, 0, stream>>>(hF, yF, g1, b1, maskF, hB);

    // ---- FFN: conv1 (K=3) + ReLU + mask ; conv2 (K=3) + mask ; LN2 ----
    gemm_tok<1, true><<<dim3(FF / 32, NPB, BB), blk128, 0, stream>>>(
        wpF1, hB, bf1, maskF, nullptr, y1B, FF, TT, DD, DD, KK,
        (long)TT * DD, (long)TT * FF, FF, 1.0f);
    gemm_tok<0, false><<<gProj, blk128, 0, stream>>>(
        wpF2, y1B, bf2, maskF, yF, nullptr, DD, TT, FF, FF, KK,
        (long)TT * FF, (long)TT * DD, DD, 1.0f);
    ln_kernel<<<gTok, blk256, 0, stream>>>(hF, yF, g2, b2, maskF, hB);
  }

  // ---- final projection (stats) + outputs ----
  pack_w<<<packGrid((long)2 * OUTC * DD), blk256, 0, stream>>>(projw, wp0, 2 * OUTC, DD, DD, 1, 0);
  gemm_tok<0, false><<<dim3(2 * OUTC / 32, NPB, BB), blk128, 0, stream>>>(
      wp0, hB, projb, maskF, yF, nullptr, 2 * OUTC, TT, DD, DD, 1,
      (long)TT * DD, (long)TT * 2 * OUTC, 2 * OUTC, 1.0f);
  final_h_kernel<<<gTok, blk256, 0, stream>>>(hF, maskF, outH);
  write_stats_kernel<<<gTok, dim3(OUTC), 0, stream>>>(yF, outM, outL);
}
